// FusedFlashLinear_6571299963235
// MI455X (gfx1250) — matrix operand order repaired and checked by reading, not compiled, not run on hardware
//
#include <hip/hip_runtime.h>

typedef __attribute__((ext_vector_type(16))) _Float16 v16h;
typedef __attribute__((ext_vector_type(8)))  float    v8f;
typedef __attribute__((ext_vector_type(4)))  _Float16 v4h;
typedef __attribute__((ext_vector_type(4)))  unsigned v4u;
typedef __attribute__((ext_vector_type(8)))  int      v8i;
typedef __attribute__((ext_vector_type(4)))  int      v4i;

#define DK 512          // D_IN  (K)
#define DN 512          // D_OUT (N)
#define BM 128          // rows per block
#define BN 128          // n-chunk width
#define ROWH (DK + 16)  // padded row length in halves (1056 B stride, 16B aligned)

// ---------------- prep: convert weight [N][K] fp32 -> f16 (same layout) -------
__global__ void wconv_f32_to_f16(const float* __restrict__ w,
                                 _Float16* __restrict__ wh) {
    int i = blockIdx.x * blockDim.x + threadIdx.x;   // one float4 per thread
    float4 f = ((const float4*)w)[i];
    v4h h = { (_Float16)f.x, (_Float16)f.y, (_Float16)f.z, (_Float16)f.w };
    ((v4h*)wh)[i] = h;
}

// ---- TDM: DMA one 128x512 f16 weight chunk global->LDS with 32B row pad -----
// Produces LDS rows of 1024B data + 32B pad = 1056B stride == ROWH halves.
__device__ __forceinline__ void tdm_load_wchunk(const _Float16* src,
                                                unsigned lds_byte_off) {
    unsigned long long ga = (unsigned long long)(uintptr_t)src;
    v4u g0;
    g0.x = 1u;                                        // count=1, user descriptor
    g0.y = lds_byte_off;                              // lds_addr [63:32]
    g0.z = (unsigned)(ga & 0xFFFFFFFFu);              // global_addr [95:64]
    g0.w = (unsigned)((ga >> 32) & 0x1FFFFFFu)        // global_addr [120:96]
         | (2u << 30);                                // type=2 ("image")
    v8i g1;
    g1[0] = (int)((1u << 16)      // data_size = 1 -> 2-byte elements
                | (1u << 20)      // pad_enable
                | (7u << 22)      // pad_interval = 256 DWORDs (one 1024B row)
                | (7u << 25));    // pad_amount  = 8 DWORDs (32B)
    g1[1] = (int)(512u << 16);    // tensor_dim0 = 512 (k extent)
    g1[2] = (int)(128u << 16);    // tensor_dim0 hi=0 | tensor_dim1 = 128 (n rows)
    g1[3] = (int)(512u << 16);    // tensor_dim1 hi=0 | tile_dim0 = 512
    g1[4] = (int)(128u);          // tile_dim1 = 128 | tile_dim2 = 0
    g1[5] = (int)(512u);          // tensor_dim0_stride = 512 (lo)
    g1[6] = 0;                    // stride0 hi | tensor_dim1_stride lo (unused, 2D)
    g1[7] = 0;
    v4i z4 = {0, 0, 0, 0};        // groups 2/3: zero (<=2D tensor)
    v8i z8 = {0, 0, 0, 0, 0, 0, 0, 0};
    // 6-arg amdgpu-toolchain (clang-23) form: (g0, g1, g2, g3, g4, cpol)
    __builtin_amdgcn_tensor_load_to_lds(g0, g1, z4, z4, z8, /*cpol=*/0);
}

// ---------------- main fused linear GEMM -------------------------------------
__global__ void __launch_bounds__(256)
fused_linear_wmma(const float* __restrict__ x,
                  const _Float16* __restrict__ wh,
                  const float* __restrict__ bias,
                  float* __restrict__ out) {
    extern __shared__ _Float16 smem[];
    _Float16* xs = smem;               // BM x ROWH halves  (x tile, f16)
    _Float16* ws = smem + BM * ROWH;   // BN x ROWH halves  (weight chunk, f16)

    const int tid  = threadIdx.x;
    const int lane = tid & 31;
    const int wv   = tid >> 5;         // wave 0..7
    const int half = lane >> 4;        // 0: lanes 0-15, 1: lanes 16-31
    const int lrow = lane & 15;

    const int m0 = blockIdx.x * BM;
    const unsigned ws_off = (unsigned)(uintptr_t)(void*)ws;

    // Kick off chunk 0's weight DMA immediately; it overlaps x staging below.
    if (wv == 0) tdm_load_wchunk(wh, ws_off);

    // ---- stage x tile: 128 x 512 fp32 -> f16 in LDS (once, full K) ----------
    // 16384 float4s total, 64 per thread; fully coalesced 16B loads.
    #pragma unroll 4
    for (int i = 0; i < 64; ++i) {
        int e   = tid + i * 256;
        int row = e >> 7;              // / (DK/4)
        int k   = (e & 127) << 2;
        float4 f = *(const float4*)(x + (size_t)(m0 + row) * DK + k);
        v4h h = { (_Float16)f.x, (_Float16)f.y, (_Float16)f.z, (_Float16)f.w };
        *(v4h*)(xs + row * ROWH + k) = h;
    }

    const _Float16* aBase = xs + (wv * 16 + lrow) * ROWH;

    // ---- loop over four N-chunks of 128 columns -----------------------------
    for (int nc = 0; nc < 4; ++nc) {
        const int n0 = nc * BN;

        if (wv == 0) __builtin_amdgcn_s_wait_tensorcnt(0);  // DMA done (wave 0)
        __syncthreads();               // publish ws to all waves; xs ready too

        // accumulators seeded with bias (C = bias broadcast down columns)
        v8f acc[8];
        #pragma unroll
        for (int t = 0; t < 8; ++t) {
            float bv = bias[n0 + t * 16 + lrow];
            acc[t] = (v8f){bv, bv, bv, bv, bv, bv, bv, bv};
        }

        // ---- K loop: 16 steps of v_wmma_f32_16x16x32_f16 per n-tile ---------
        #pragma unroll 2
        for (int k0 = 0; k0 < DK; k0 += 32) {
            // A fragment (ISA 16-bit A 16x32 layout):
            // lanes 0-15: VGPR0-3 = K[k0..k0+7],  VGPR4-7 = K[k0+16..k0+23]
            // lanes16-31: VGPR0-3 = K[k0+8..+15], VGPR4-7 = K[k0+24..+31]
            union { uint4 q[2]; v16h v; } a;
            a.q[0] = *(const uint4*)(aBase + k0 + half * 8);
            a.q[1] = *(const uint4*)(aBase + k0 + 16 + half * 8);

            #pragma unroll
            for (int t = 0; t < 8; ++t) {
                // B fragment: lane = column n, K[k0+8h..+7] and K[k0+16+8h..+7] per half-wave
                const _Float16* bB = ws + (t * 16 + lrow) * ROWH + half * 8 + k0;
                union { uint4 q[2]; v16h v; } b;
                b.q[0] = *(const uint4*)(bB);
                b.q[1] = *(const uint4*)(bB + 16);

                acc[t] = __builtin_amdgcn_wmma_f32_16x16x32_f16(
                    /*neg_a=*/false, a.v, /*neg_b=*/false, b.v,
                    /*c_mod=*/(short)0, acc[t],
                    /*reuse_a=*/false, /*reuse_b=*/false);
            }
        }

        __syncthreads();               // all waves done reading ws[nc]
        if (nc < 3 && wv == 0)         // start next chunk's DMA ASAP
            tdm_load_wchunk(wh + (size_t)(n0 + BN) * DK, ws_off);

        // ---- store D: lane l, VGPR v -> (M = v + 8*half, N = lrow) ----------
        // (overlaps with the next chunk's DMA)
        #pragma unroll
        for (int t = 0; t < 8; ++t) {
            const int n = n0 + t * 16 + lrow;
            #pragma unroll
            for (int v = 0; v < 8; ++v) {
                int m = m0 + wv * 16 + v + half * 8;
                out[(size_t)m * DN + n] = acc[t][v];
            }
        }
    }
}

extern "C" void kernel_launch(void* const* d_in, const int* in_sizes, int n_in,
                              void* d_out, int out_size, void* d_ws, size_t ws_size,
                              hipStream_t stream) {
    const float* x    = (const float*)d_in[0];   // [32, 4096, 512]
    const float* w    = (const float*)d_in[1];   // [512, 512]
    const float* bias = (const float*)d_in[2];   // [512]
    float* out        = (float*)d_out;           // [32, 4096, 512]
    _Float16* wh      = (_Float16*)d_ws;         // 512*512 f16 = 512 KB scratch

    // 1) convert weight fp32 -> f16 (keeps [n][k] layout)
    wconv_f32_to_f16<<<(DN * DK / 4) / 256, 256, 0, stream>>>(w, wh);

    // 2) main GEMM: 131072/128 = 1024 blocks, 256 threads (8 wave32),
    //    264 KB dynamic LDS (needs CDNA5's 320 KB WGP LDS)
    const size_t smem_bytes = (size_t)(BM + BN) * ROWH * sizeof(_Float16);
    fused_linear_wmma<<<131072 / BM, 256, smem_bytes, stream>>>(x, wh, bias, out);
}
